// GeodesicBranch_30803505446966
// MI455X (gfx1250) — matrix operand order repaired and checked by reading, not compiled, not run on hardware
//
#include <hip/hip_runtime.h>
#include <hip/hip_bf16.h>
#include <math.h>

// ---------------------------------------------------------------------------
// CDNA5 / gfx1250 implementation.
// All dense GEMMs run on v_wmma_f32_16x16x32_bf16 (bf16 in, f32 accumulate).
// The GEMM k-loop is software-pipelined with two fragment buffers and
// sched_barriers so loads of step i+1 overlap the 8 WMMAs of step i
// (staged s_wait_loadcnt instead of waiting to 0 before every WMMA).
// Sparse scatter uses global_atomic_add_f32; segment-max uses ordered-uint
// atomicMax. Wave32 layouts used here (contiguous k order in A and B):
//   A 16x32 bf16 : lane L -> row = L%16 ; elems 0..15 = K kb..kb+15,
//                  kb = (L>=16)?16:0
//   B 32x16 bf16 : lane L -> col = L%16 ; elem e -> K = ((L>=16)?16:0)+e
//   C/D 16x16 f32: VGPR i -> row = i + ((L>=16)?8:0), col = L%16
// ---------------------------------------------------------------------------

typedef __bf16 bf16_t;
typedef bf16_t v16bf __attribute__((ext_vector_type(16)));
typedef bf16_t v8bf  __attribute__((ext_vector_type(8)));
typedef float  v8f   __attribute__((ext_vector_type(8)));

union AV { v8bf h[2]; v16bf f; };

static inline long cdiv(long a, long b) { return (a + b - 1) / b; }
// packed-B tile count: ceil(N/16) rounded up to a multiple of 8 (one 128-wide
// N-block), so every workgroup computes a full 8-tile stripe branch-free.
static inline int ntiles_padded(int N) { return (int)cdiv(N, 128) * 8; }

// ------------------------------ elementwise --------------------------------

__global__ void k_fill_f32(float* __restrict__ p, float v, long n) {
  long i = (long)blockIdx.x * blockDim.x + threadIdx.x;
  if (i < n) p[i] = v;
}

__global__ void k_fill_u32(unsigned* __restrict__ p, unsigned v, long n) {
  long i = (long)blockIdx.x * blockDim.x + threadIdx.x;
  if (i < n) p[i] = v;
}

// f32 [M,ld] -> bf16 [M,ldp], zero-padding columns K..ldp
__global__ void k_cvt_bf16(const float* __restrict__ X, int ld,
                           bf16_t* __restrict__ Y, int ldp, long M, int K) {
  long i = (long)blockIdx.x * blockDim.x + threadIdx.x;
  long total = M * (long)ldp;
  if (i >= total) return;
  int col = (int)(i % ldp);
  long row = i / ldp;
  float v = (col < K) ? X[row * (long)ld + col] : 0.f;
  Y[i] = (bf16_t)v;
}

// evecs [V,K] f32 -> evecsT [K,Vpad] bf16 (zero pad v >= V)
__global__ void k_cvt_transpose_bf16(const float* __restrict__ X,
                                     bf16_t* __restrict__ Y,
                                     int V, int K, int Vpad) {
  long i = (long)blockIdx.x * blockDim.x + threadIdx.x;
  long total = (long)K * Vpad;
  if (i >= total) return;
  int v = (int)(i % Vpad);
  int k = (int)(i / Vpad);
  float val = (v < V) ? X[(long)v * K + k] : 0.f;
  Y[i] = (bf16_t)val;
}

// Pack B[K,N] f32 (opt row-scaled) into per-lane WMMA B-fragment layout:
// Bp[tk][tn][lane][e], 512 bf16 per 32x16 tile. Ntiles is padded to a
// multiple of 8; out-of-range K/N entries are zero-filled.
__global__ void k_pack_b(const float* __restrict__ B, int ldb,
                         const float* __restrict__ rowScale,
                         bf16_t* __restrict__ Bp, long K, int N,
                         int Ntiles, long total) {
  long i = (long)blockIdx.x * blockDim.x + threadIdx.x;
  if (i >= total) return;
  int e = (int)(i & 15);
  int lane = (int)((i >> 4) & 31);
  long tile = i >> 9;
  int tn = (int)(tile % Ntiles);
  long tk = tile / Ntiles;
  int col = tn * 16 + (lane & 15);
  long k = tk * 32 + ((lane >= 16) ? 16 : 0) + e;
  float v = 0.f;
  if (k < K && col < N) {
    v = B[k * (long)ldb + col];
    if (rowScale) v *= rowScale[k];
  }
  Bp[i] = (bf16_t)v;
}

// ------------------------------ WMMA GEMM ----------------------------------
// C[M,N] = act(A[M,Kp]_bf16 @ B_packed + bias (+ resid)), or split-K atomic.
// 256 threads = 8 waves; block tile 128x128; wave tile 16 x 128.
// Double-buffered fragments: loads for k-step i+1 are issued before the 8
// WMMAs of step i; sched_barrier(0) pins the order so RA keeps both buffers.
__global__ void __launch_bounds__(256)
k_gemm_wmma(const bf16_t* __restrict__ A, int lda,
            const bf16_t* __restrict__ Bp, int ntilesN,
            const float* __restrict__ bias,
            const float* __restrict__ resid,
            float* __restrict__ C,
            int M, int N, int Kp, int kSlice,
            int relu, int atomic) {
  const int wave = threadIdx.x >> 5;
  const int lane = threadIdx.x & 31;
  const int hi = lane >> 4;
  const int l16 = lane & 15;
  const int mrow0 = blockIdx.x * 128 + wave * 16;
  const int nblk0 = blockIdx.y * 128;

  int kbeg = blockIdx.z * kSlice;
  int kend = kbeg + kSlice;
  if (kend > Kp) kend = Kp;
  if (kbeg >= kend) return;

  v8f zero = {0.f, 0.f, 0.f, 0.f, 0.f, 0.f, 0.f, 0.f};
  v8f acc[8];
#pragma unroll
  for (int t = 0; t < 8; ++t) acc[t] = zero;

  int rowA = mrow0 + l16;
  if (rowA >= M) rowA = M - 1;  // clamp; garbage rows are never stored
  const int kbase = hi ? 16 : 0;

  // strength-reduced fragment pointers
  const bf16_t* aptr = A + (long)rowA * lda + kbeg + kbase;
  const bf16_t* bptr =
      Bp + ((((long)(kbeg >> 5)) * ntilesN + (nblk0 >> 4)) << 9) + (lane << 4);
  const long bstep = (long)ntilesN << 9;  // bytes/2 per k-tile of packed B

  AV a0, a1;
  v16bf b0[8], b1[8];

  auto loadf = [&](AV& av_, v16bf* bv_, const bf16_t* ap, const bf16_t* bp) {
    av_.h[0] = *(const v8bf*)ap;
    av_.h[1] = *(const v8bf*)(ap + 8);
#pragma unroll
    for (int t = 0; t < 8; ++t) bv_[t] = *(const v16bf*)(bp + ((long)t << 9));
  };
  auto computef = [&](const AV& av_, const v16bf* bv_) {
#pragma unroll
    for (int t = 0; t < 8; ++t)
      acc[t] = __builtin_amdgcn_wmma_f32_16x16x32_bf16(
          false, av_.f, false, bv_[t], (short)0, acc[t], false, false);
  };

  const int nk = (kend - kbeg) >> 5;  // all Kp/kSlice are multiples of 32
  loadf(a0, b0, aptr, bptr);
  aptr += 32;
  bptr += bstep;
  for (int r = nk; r >= 2; r -= 2) {
    loadf(a1, b1, aptr, bptr);
    aptr += 32;
    bptr += bstep;
    __builtin_amdgcn_sched_barrier(0);
    computef(a0, b0);
    if (r >= 3) {
      loadf(a0, b0, aptr, bptr);
      aptr += 32;
      bptr += bstep;
    }
    __builtin_amdgcn_sched_barrier(0);
    computef(a1, b1);
  }
  if (nk & 1) computef(a0, b0);

#pragma unroll
  for (int t = 0; t < 8; ++t) {
    int col = nblk0 + t * 16 + l16;
    if (col >= N) continue;
#pragma unroll
    for (int i = 0; i < 8; ++i) {
      int row = mrow0 + i + (hi ? 8 : 0);
      if (row >= M) continue;
      float v = acc[t][i];
      long off = (long)row * N + col;
      if (atomic) {
        atomicAdd(&C[off], v);
      } else {
        if (bias) v += bias[col];
        if (resid) v += resid[off];
        if (relu) v = fmaxf(v, 0.f);
        C[off] = v;
      }
    }
  }
}

// ------------------------------ sparse ops ---------------------------------

// Out[idx[e].0, colOff + c] += val[e] * X[idx[e].1, c]  (f32 atomics)
__global__ void k_spmm_scatter(const int* __restrict__ idx,
                               const float* __restrict__ val,
                               const float* __restrict__ X, int C,
                               float* __restrict__ Out, int ldo, int colOff,
                               long E) {
  long i = (long)blockIdx.x * blockDim.x + threadIdx.x;
  int cq = C >> 2;
  long e = i / cq;
  if (e >= E) return;
  int c4 = (int)(i % cq) << 2;
  int dst = idx[2 * e];
  int src = idx[2 * e + 1];
  float v = val[e];
  float4 x = *(const float4*)(X + (long)src * C + c4);
  float* o = Out + (long)dst * ldo + colOff + c4;
  atomicAdd(o + 0, v * x.x);
  atomicAdd(o + 1, v * x.y);
  atomicAdd(o + 2, v * x.z);
  atomicAdd(o + 3, v * x.w);
}

__device__ inline unsigned f32_ord(float f) {
  unsigned b = __float_as_uint(f);
  return (b & 0x80000000u) ? ~b : (b | 0x80000000u);
}
__device__ inline float ord_f32(unsigned k) {
  return __uint_as_float((k & 0x80000000u) ? (k & 0x7fffffffu) : ~k);
}

__global__ void k_segmax_scatter(const float* __restrict__ x,
                                 const int* __restrict__ tr,
                                 unsigned* __restrict__ keys, int C, long V) {
  long i = (long)blockIdx.x * blockDim.x + threadIdx.x;
  long total = V * (long)C;
  if (i >= total) return;
  int c = (int)(i % C);
  long v = i / C;
  atomicMax(&keys[(long)tr[v] * C + c], f32_ord(x[i]));
}

__global__ void k_segmax_decode(const unsigned* __restrict__ keys,
                                float* __restrict__ xm, long n) {
  long i = (long)blockIdx.x * blockDim.x + threadIdx.x;
  if (i >= n) return;
  float f = ord_f32(keys[i]);
  if (f == -INFINITY) f = 0.f;  // reference: where(isneginf, 0)
  xm[i] = f;
}

__global__ void k_gather_rows(const float* __restrict__ xm,
                              const int* __restrict__ tr,
                              float* __restrict__ y, int C, long V) {
  long i = (long)blockIdx.x * blockDim.x + threadIdx.x;
  long total = V * (long)C;
  if (i >= total) return;
  int c = (int)(i % C);
  long v = i / C;
  y[i] = xm[(long)tr[v] * C + c];
}

__global__ void k_copy_cols(const float* __restrict__ src, int lds,
                            float* __restrict__ dst, int ldd, int off, int C,
                            long V) {
  long i = (long)blockIdx.x * blockDim.x + threadIdx.x;
  long total = V * (long)C;
  if (i >= total) return;
  int c = (int)(i % C);
  long v = i / C;
  dst[v * (long)ldd + off + c] = src[v * (long)lds + c];
}

// spec2[k,c] = exp(-evals[k]*max(dt[c],1e-8)) * spec[k,c]
__global__ void k_coef(const float* __restrict__ spec,
                       const float* __restrict__ evals,
                       const float* __restrict__ dt,
                       float* __restrict__ spec2, int K, int C) {
  long i = (long)blockIdx.x * blockDim.x + threadIdx.x;
  long total = (long)K * C;
  if (i >= total) return;
  int c = (int)(i % C);
  int k = (int)(i / C);
  spec2[i] = spec[i] * expf(-evals[k] * fmaxf(dt[c], 1e-8f));
}

// W[2C,2C] = [[Are, Aim], [-Aim, Are]]
__global__ void k_build_W(const float* __restrict__ Are,
                          const float* __restrict__ Aim,
                          float* __restrict__ W, int C) {
  long i = (long)blockIdx.x * blockDim.x + threadIdx.x;
  int tc = 2 * C;
  long total = (long)tc * tc;
  if (i >= total) return;
  int col = (int)(i % tc);
  int row = (int)(i / tc);
  float v;
  if (row < C) v = (col < C) ? Are[(long)row * C + col] : Aim[(long)row * C + col - C];
  else {
    int r = row - C;
    v = (col < C) ? -Aim[(long)r * C + col] : Are[(long)r * C + col - C];
  }
  W[i] = v;
}

// h = [x, xd, tanh(gX*Br + gY*Bi)]
__global__ void k_build_h(const float* __restrict__ x,
                          const float* __restrict__ xd,
                          const float* __restrict__ gXY,
                          const float* __restrict__ BrBi,
                          float* __restrict__ h, int C, long V) {
  long i = (long)blockIdx.x * blockDim.x + threadIdx.x;
  long total = V * (long)C;
  if (i >= total) return;
  int c = (int)(i % C);
  long v = i / C;
  long b2 = v * (long)(2 * C);
  float gX = gXY[b2 + c], gY = gXY[b2 + C + c];
  float Br = BrBi[b2 + c], Bi = BrBi[b2 + C + c];
  long b3 = v * (long)(3 * C);
  h[b3 + c] = x[v * (long)C + c];
  h[b3 + C + c] = xd[v * (long)C + c];
  h[b3 + 2 * C + c] = tanhf(gX * Br + gY * Bi);
}

// ------------------------------ host helpers -------------------------------

struct Scratch {
  bf16_t *Bp, *abf, *hbf;
  float *spec, *spec2, *Wbuf, *xd, *gXY, *BrBi, *h, *t1;
  unsigned* keys;
};

static void do_pack_b(hipStream_t s, const float* B, int ldb, const float* scale,
                      bf16_t* Bp, long K, int N) {
  int Ntiles = ntiles_padded(N);
  long Ktiles = (K + 31) / 32;
  long total = Ktiles * Ntiles * 512;
  k_pack_b<<<(unsigned)cdiv(total, 256), 256, 0, s>>>(B, ldb, scale, Bp, K, N,
                                                      Ntiles, total);
}

static void do_cvt(hipStream_t s, const float* X, int ld, bf16_t* Y, int ldp,
                   long M, int K) {
  long total = M * (long)ldp;
  k_cvt_bf16<<<(unsigned)cdiv(total, 256), 256, 0, s>>>(X, ld, Y, ldp, M, K);
}

static void do_gemm(hipStream_t s, const bf16_t* A, int lda, const bf16_t* Bp,
                    int N, const float* bias, const float* resid, float* C,
                    long M, int Kp, bool relu, int splitK) {
  int ntilesN = ntiles_padded(N);
  dim3 grid((unsigned)cdiv(M, 128), (unsigned)cdiv(N, 128), (unsigned)splitK);
  int ktiles = Kp / 32;
  int kt = (ktiles + splitK - 1) / splitK;
  int kSlice = kt * 32;
  k_gemm_wmma<<<grid, 256, 0, s>>>(A, lda, Bp, ntilesN, bias, resid, C, (int)M,
                                   N, Kp, kSlice, relu ? 1 : 0,
                                   splitK > 1 ? 1 : 0);
}

static void do_zero(hipStream_t s, float* p, long n) {
  k_fill_f32<<<(unsigned)cdiv(n, 256), 256, 0, s>>>(p, 0.f, n);
}

// One diffusion/gating/MLP block (reference _block).
static void run_block(hipStream_t s, int V, int Vpad, int C,
                      const float* xin, float* xout,
                      const float* mass, const float* evals,
                      const bf16_t* evecs_bf, const bf16_t* evecsT_bf,
                      const int* gxi, const float* gxv,
                      const int* gyi, const float* gyv, long E,
                      const float* dt, const float* Are, const float* Aim,
                      const float* w0, const float* b0,
                      const float* w1, const float* b1,
                      const float* w2, const float* b2, Scratch& sc) {
  const int K = 128;
  // spectral projection: spec[K,C] = evecsT @ (x * mass), split-K + atomics
  do_pack_b(s, xin, C, mass, sc.Bp, V, C);
  do_zero(s, sc.spec, (long)K * C);
  do_gemm(s, evecsT_bf, Vpad, sc.Bp, C, nullptr, nullptr, sc.spec, K, Vpad,
          false, 128);
  // heat-kernel coefficients
  k_coef<<<(unsigned)cdiv((long)K * C, 256), 256, 0, s>>>(sc.spec, evals, dt,
                                                          sc.spec2, K, C);
  // xd = evecs @ spec2
  do_pack_b(s, sc.spec2, C, nullptr, sc.Bp, K, C);
  do_gemm(s, evecs_bf, K, sc.Bp, C, nullptr, nullptr, sc.xd, V, K, false, 1);
  // gX | gY via sparse scatter
  do_zero(s, sc.gXY, (long)V * 2 * C);
  long tE = E * (long)(C / 4);
  k_spmm_scatter<<<(unsigned)cdiv(tE, 256), 256, 0, s>>>(gxi, gxv, sc.xd, C,
                                                         sc.gXY, 2 * C, 0, E);
  k_spmm_scatter<<<(unsigned)cdiv(tE, 256), 256, 0, s>>>(gyi, gyv, sc.xd, C,
                                                         sc.gXY, 2 * C, C, E);
  // [Br Bi] = [gX gY] @ [[Are,Aim],[-Aim,Are]]
  k_build_W<<<(unsigned)cdiv((long)4 * C * C, 256), 256, 0, s>>>(Are, Aim,
                                                                 sc.Wbuf, C);
  do_pack_b(s, sc.Wbuf, 2 * C, nullptr, sc.Bp, 2 * C, 2 * C);
  do_cvt(s, sc.gXY, 2 * C, sc.abf, 2 * C, V, 2 * C);
  do_gemm(s, sc.abf, 2 * C, sc.Bp, 2 * C, nullptr, nullptr, sc.BrBi, V, 2 * C,
          false, 1);
  // h = [x, xd, tanh(gX*Br+gY*Bi)]
  k_build_h<<<(unsigned)cdiv((long)V * C, 256), 256, 0, s>>>(xin, sc.xd, sc.gXY,
                                                             sc.BrBi, sc.h, C, V);
  do_cvt(s, sc.h, 3 * C, sc.hbf, 3 * C, V, 3 * C);
  // MLP: relu(h@w0+b0) -> relu(@w1+b1) -> @w2+b2 + x
  do_pack_b(s, w0, C, nullptr, sc.Bp, 3 * C, C);
  do_gemm(s, sc.hbf, 3 * C, sc.Bp, C, b0, nullptr, sc.t1, V, 3 * C, true, 1);
  do_cvt(s, sc.t1, C, sc.abf, C, V, C);
  do_pack_b(s, w1, C, nullptr, sc.Bp, C, C);
  do_gemm(s, sc.abf, C, sc.Bp, C, b1, nullptr, sc.t1, V, C, true, 1);
  do_cvt(s, sc.t1, C, sc.abf, C, V, C);
  do_pack_b(s, w2, C, nullptr, sc.Bp, C, C);
  do_gemm(s, sc.abf, C, sc.Bp, C, b2, xin, xout, V, C, false, 1);
}

// ------------------------------ entry point --------------------------------

extern "C" void kernel_launch(void* const* d_in, const int* in_sizes, int n_in,
                              void* d_out, int out_size, void* d_ws,
                              size_t ws_size, hipStream_t stream) {
  (void)in_sizes; (void)n_in; (void)out_size; (void)ws_size;
  const int V3 = 100000, Vm = 25000, K = 128;
  const long E3 = 800000, Em = 200000;
  const int Vpad3 = 100000;                 // multiple of 32
  const int Vpadm = ((Vm + 31) / 32) * 32;  // 25024

  const float* x_in   = (const float*)d_in[0];
  const float* mass3  = (const float*)d_in[1];
  const float* evals3 = (const float*)d_in[2];
  const float* evecs3 = (const float*)d_in[3];
  const int*   gX3i   = (const int*)d_in[4];
  const float* gX3v   = (const float*)d_in[5];
  const int*   gY3i   = (const int*)d_in[6];
  const float* gY3v   = (const float*)d_in[7];
  const float* massm  = (const float*)d_in[8];
  const float* evalsm = (const float*)d_in[9];
  const float* evecsm = (const float*)d_in[10];
  const int*   gXmi   = (const int*)d_in[11];
  const float* gXmv   = (const float*)d_in[12];
  const int*   gYmi   = (const int*)d_in[13];
  const float* gYmv   = (const float*)d_in[14];
  const int*   traces = (const int*)d_in[15];
  const float* in_w   = (const float*)d_in[16];
  const float* in_b   = (const float*)d_in[17];
  const float* widen_w  = (const float*)d_in[18];
  const float* widen_b  = (const float*)d_in[19];
  const float* narrow_w = (const float*)d_in[20];
  const float* narrow_b = (const float*)d_in[21];
  const float* halve_w  = (const float*)d_in[22];
  const float* halve_b  = (const float*)d_in[23];
  const float* out_w  = (const float*)d_in[24];
  const float* out_b  = (const float*)d_in[25];
  const float* enc[9], *mid[9], *dec[9];
  for (int j = 0; j < 9; ++j) enc[j] = (const float*)d_in[26 + j];
  for (int j = 0; j < 9; ++j) mid[j] = (const float*)d_in[35 + j];
  for (int j = 0; j < 9; ++j) dec[j] = (const float*)d_in[44 + j];

  // bump allocator over d_ws
  char* wp = (char*)d_ws;
  auto alloc = [&](size_t bytes) -> char* {
    char* r = wp;
    wp += (bytes + 255) & ~(size_t)255;
    return r;
  };
  float* bufA = (float*)alloc((size_t)V3 * 128 * 4);  // x ping/pong + xe
  float* bufB = (float*)alloc((size_t)V3 * 128 * 4);
  float* bufC = (float*)alloc((size_t)V3 * 128 * 4);
  Scratch sc;
  const size_t MAXVC  = (size_t)V3 * 128;
  sc.xd   = (float*)alloc(MAXVC * 4);                  // max(V3*128, Vm*256)
  sc.gXY  = (float*)alloc(MAXVC * 2 * 4);              // max(V3*256, Vm*512)
  sc.BrBi = (float*)alloc(MAXVC * 2 * 4);
  sc.h    = (float*)alloc(MAXVC * 3 * 4);              // max(V3*384, Vm*768)
  sc.t1   = (float*)alloc(MAXVC * 4);
  sc.hbf  = (bf16_t*)alloc(MAXVC * 3 * 2);
  sc.abf  = (bf16_t*)alloc(MAXVC * 2 * 2);
  sc.Bp   = (bf16_t*)alloc((size_t)Vpad3 * 128 * 2);   // biggest packed B
  sc.spec  = (float*)alloc((size_t)K * 256 * 4);
  sc.spec2 = (float*)alloc((size_t)K * 256 * 4);
  sc.Wbuf  = (float*)alloc((size_t)512 * 512 * 4);
  sc.keys  = (unsigned*)alloc((size_t)Vm * 128 * 4);
  bf16_t* ev3  = (bf16_t*)alloc((size_t)V3 * 128 * 2);
  bf16_t* ev3T = (bf16_t*)alloc((size_t)128 * Vpad3 * 2);
  bf16_t* evm  = (bf16_t*)alloc((size_t)Vm * 128 * 2);
  bf16_t* evmT = (bf16_t*)alloc((size_t)128 * Vpadm * 2);

  hipStream_t s = stream;

  // one-time (per launch) bf16 conversions of eigenbases
  do_cvt(s, evecs3, 128, ev3, 128, V3, 128);
  k_cvt_transpose_bf16<<<(unsigned)cdiv((long)128 * Vpad3, 256), 256, 0, s>>>(
      evecs3, ev3T, V3, 128, Vpad3);
  do_cvt(s, evecsm, 128, evm, 128, Vm, 128);
  k_cvt_transpose_bf16<<<(unsigned)cdiv((long)128 * Vpadm, 256), 256, 0, s>>>(
      evecsm, evmT, Vm, 128, Vpadm);

  // input projection: x3 = relu(x_in @ in_w + in_b)   (K=16 padded to 32)
  do_cvt(s, x_in, 16, sc.abf, 32, V3, 16);
  do_pack_b(s, in_w, 128, nullptr, sc.Bp, 16, 128);
  do_gemm(s, sc.abf, 32, sc.Bp, 128, in_b, nullptr, bufA, V3, 32, true, 1);

  // encoder blocks (C=128): A -> B -> A   (xe := bufA afterwards)
  run_block(s, V3, Vpad3, 128, bufA, bufB, mass3, evals3, ev3, ev3T, gX3i, gX3v,
            gY3i, gY3v, E3, enc[0] + 0 * 128, enc[1] + 0 * 128 * 128,
            enc[2] + 0 * 128 * 128, enc[3] + 0 * 384 * 128, enc[4] + 0 * 128,
            enc[5] + 0 * 128 * 128, enc[6] + 0 * 128, enc[7] + 0 * 128 * 128,
            enc[8] + 0 * 128, sc);
  run_block(s, V3, Vpad3, 128, bufB, bufA, mass3, evals3, ev3, ev3T, gX3i, gX3v,
            gY3i, gY3v, E3, enc[0] + 1 * 128, enc[1] + 1 * 128 * 128,
            enc[2] + 1 * 128 * 128, enc[3] + 1 * 384 * 128, enc[4] + 1 * 128,
            enc[5] + 1 * 128 * 128, enc[6] + 1 * 128, enc[7] + 1 * 128 * 128,
            enc[8] + 1 * 128, sc);

  // segment-max pool V3 -> Vm, -inf -> 0, then widen to C=256
  k_fill_u32<<<(unsigned)cdiv((long)Vm * 128, 256), 256, 0, s>>>(
      sc.keys, 0x007FFFFFu /* ord(-inf) */, (long)Vm * 128);
  k_segmax_scatter<<<(unsigned)cdiv((long)V3 * 128, 256), 256, 0, s>>>(
      bufA, traces, sc.keys, 128, V3);
  k_segmax_decode<<<(unsigned)cdiv((long)Vm * 128, 256), 256, 0, s>>>(
      sc.keys, sc.t1, (long)Vm * 128);
  do_cvt(s, sc.t1, 128, sc.abf, 128, Vm, 128);
  do_pack_b(s, widen_w, 256, nullptr, sc.Bp, 128, 256);
  do_gemm(s, sc.abf, 128, sc.Bp, 256, widen_b, nullptr, bufB, Vm, 128, true, 1);

  // middle blocks (C=256): B -> C -> B
  run_block(s, Vm, Vpadm, 256, bufB, bufC, massm, evalsm, evm, evmT, gXmi, gXmv,
            gYmi, gYmv, Em, mid[0] + 0 * 256, mid[1] + 0 * 256 * 256,
            mid[2] + 0 * 256 * 256, mid[3] + 0 * 768 * 256, mid[4] + 0 * 256,
            mid[5] + 0 * 256 * 256, mid[6] + 0 * 256, mid[7] + 0 * 256 * 256,
            mid[8] + 0 * 256, sc);
  run_block(s, Vm, Vpadm, 256, bufC, bufB, massm, evalsm, evm, evmT, gXmi, gXmv,
            gYmi, gYmv, Em, mid[0] + 1 * 256, mid[1] + 1 * 256 * 256,
            mid[2] + 1 * 256 * 256, mid[3] + 1 * 768 * 256, mid[4] + 1 * 256,
            mid[5] + 1 * 256 * 256, mid[6] + 1 * 256, mid[7] + 1 * 256 * 256,
            mid[8] + 1 * 256, sc);

  // unpool: gather, narrow, concat with xe, halve
  k_gather_rows<<<(unsigned)cdiv((long)V3 * 256, 256), 256, 0, s>>>(
      bufB, traces, sc.gXY, 256, V3);
  do_cvt(s, sc.gXY, 256, sc.abf, 256, V3, 256);
  do_pack_b(s, narrow_w, 128, nullptr, sc.Bp, 256, 128);
  do_gemm(s, sc.abf, 256, sc.Bp, 128, narrow_b, nullptr, sc.BrBi, V3, 256, true, 1);
  k_copy_cols<<<(unsigned)cdiv((long)V3 * 128, 256), 256, 0, s>>>(
      sc.BrBi, 128, sc.h, 256, 0, 128, V3);
  k_copy_cols<<<(unsigned)cdiv((long)V3 * 128, 256), 256, 0, s>>>(
      bufA, 128, sc.h, 256, 128, 128, V3);
  do_cvt(s, sc.h, 256, sc.abf, 256, V3, 256);
  do_pack_b(s, halve_w, 128, nullptr, sc.Bp, 256, 128);
  do_gemm(s, sc.abf, 256, sc.Bp, 128, halve_b, nullptr, bufC, V3, 256, true, 1);

  // decoder blocks (C=128): C -> B -> C
  run_block(s, V3, Vpad3, 128, bufC, bufB, mass3, evals3, ev3, ev3T, gX3i, gX3v,
            gY3i, gY3v, E3, dec[0] + 0 * 128, dec[1] + 0 * 128 * 128,
            dec[2] + 0 * 128 * 128, dec[3] + 0 * 384 * 128, dec[4] + 0 * 128,
            dec[5] + 0 * 128 * 128, dec[6] + 0 * 128, dec[7] + 0 * 128 * 128,
            dec[8] + 0 * 128, sc);
  run_block(s, V3, Vpad3, 128, bufB, bufC, mass3, evals3, ev3, ev3T, gX3i, gX3v,
            gY3i, gY3v, E3, dec[0] + 1 * 128, dec[1] + 1 * 128 * 128,
            dec[2] + 1 * 128 * 128, dec[3] + 1 * 384 * 128, dec[4] + 1 * 128,
            dec[5] + 1 * 128 * 128, dec[6] + 1 * 128, dec[7] + 1 * 128 * 128,
            dec[8] + 1 * 128, sc);

  // output head: y = y3 @ out_w + out_b  -> d_out [V3, 21]
  do_cvt(s, bufC, 128, sc.abf, 128, V3, 128);
  do_pack_b(s, out_w, 21, nullptr, sc.Bp, 128, 21);
  do_gemm(s, sc.abf, 128, sc.Bp, 21, out_b, nullptr, (float*)d_out, V3, 128,
          false, 1);
}
